// SaccadicBlock_51780125720827
// MI455X (gfx1250) — matrix operand order repaired and checked by reading, not compiled, not run on hardware
//
#include <hip/hip_runtime.h>

// ---------------------------------------------------------------------------
// Types for CDNA5 WMMA
// ---------------------------------------------------------------------------
typedef __attribute__((ext_vector_type(16))) __bf16 v16bf;
typedef __attribute__((ext_vector_type(8)))  float  v8f;

#if defined(__gfx1250__)
#define USE_ASYNC_LDS 1
#endif

__device__ __forceinline__ __bf16 f2bf(float f) {
    unsigned u = __float_as_uint(f);
    u = (u + 0x7FFFu + ((u >> 16) & 1u)) >> 16;
    unsigned short s = (unsigned short)u;
    __bf16 r;
    __builtin_memcpy(&r, &s, 2);
    return r;
}

__device__ __forceinline__ float gelu_f(float x) {
    return 0.5f * x * (1.0f + erff(x * 0.70710678118654752f));
}

// ---------------------------------------------------------------------------
// Weight convert + transpose: W [Nout,K] f32  ->  Wt [K,Nout] bf16
// ---------------------------------------------------------------------------
__global__ __launch_bounds__(256) void k_w2bf(const float* __restrict__ W,
                                              __bf16* __restrict__ Wt,
                                              int Nout, int K) {
    size_t i = (size_t)blockIdx.x * 256 + threadIdx.x;
    if (i >= (size_t)Nout * K) return;
    int n = (int)(i % Nout);
    int k = (int)(i / Nout);
    Wt[i] = f2bf(W[(size_t)n * K + k]);
}

// ---------------------------------------------------------------------------
// LayerNorm over D=768 rows (one workgroup per row), optional positional add
// ---------------------------------------------------------------------------
__global__ __launch_bounds__(256) void k_ln(const float* __restrict__ in,
                                            float* __restrict__ out,
                                            const float* __restrict__ g,
                                            const float* __restrict__ b,
                                            const float* __restrict__ pos,
                                            int posmod) {
    int row = blockIdx.x, tid = threadIdx.x;
    const float* x = in + (size_t)row * 768;
    float v0 = x[tid], v1 = x[tid + 256], v2 = x[tid + 512];
    __shared__ float red[256];
    red[tid] = v0 + v1 + v2;
    __syncthreads();
    for (int s = 128; s > 0; s >>= 1) { if (tid < s) red[tid] += red[tid + s]; __syncthreads(); }
    float mu = red[0] * (1.0f / 768.0f);
    __syncthreads();
    red[tid] = v0 * v0 + v1 * v1 + v2 * v2;
    __syncthreads();
    for (int s = 128; s > 0; s >>= 1) { if (tid < s) red[tid] += red[tid + s]; __syncthreads(); }
    float var = red[0] * (1.0f / 768.0f) - mu * mu;
    float rs = rsqrtf(var + 1e-5f);
    float* o = out + (size_t)row * 768;
    const float* pp = pos ? pos + (size_t)(row % posmod) * 768 : nullptr;
    o[tid]       = (v0 - mu) * rs * g[tid]       + b[tid]       + (pp ? pp[tid]       : 0.0f);
    o[tid + 256] = (v1 - mu) * rs * g[tid + 256] + b[tid + 256] + (pp ? pp[tid + 256] : 0.0f);
    o[tid + 512] = (v2 - mu) * rs * g[tid + 512] + b[tid + 512] + (pp ? pp[tid + 512] : 0.0f);
}

// ---------------------------------------------------------------------------
// Pooling stats per block of 8 rows: softmax-weighted mean / std / max
// P3 row = [mn(768) | sd(768) | mx(768)]
// ---------------------------------------------------------------------------
__global__ __launch_bounds__(256) void k_pool(const float* __restrict__ h,
                                              const float* __restrict__ ww,
                                              const float* __restrict__ wb,
                                              float* __restrict__ P3) {
    int blk = blockIdx.x;  // b*512 + n
    int tid = threadIdx.x;
    const float* base = h + (size_t)blk * 8 * 768;
    __shared__ float red[256];
    __shared__ float wt[8];
    for (int j = 0; j < 8; ++j) {
        float s = 0.0f;
        for (int d = tid; d < 768; d += 256) s += base[j * 768 + d] * ww[d];
        red[tid] = s;
        __syncthreads();
        for (int st = 128; st > 0; st >>= 1) { if (tid < st) red[tid] += red[tid + st]; __syncthreads(); }
        if (tid == 0) wt[j] = red[0] + wb[0];
        __syncthreads();
    }
    if (tid == 0) {
        float mx = wt[0];
        for (int j = 1; j < 8; ++j) mx = fmaxf(mx, wt[j]);
        float sum = 0.0f;
        for (int j = 0; j < 8; ++j) { float e = expf(wt[j] - mx); wt[j] = e; sum += e; }
        float inv = 1.0f / sum;
        for (int j = 0; j < 8; ++j) wt[j] *= inv;
    }
    __syncthreads();
    size_t orow = (size_t)blk * 2304;
    for (int d = tid; d < 768; d += 256) {
        float v[8], mn = 0.0f;
        for (int j = 0; j < 8; ++j) { v[j] = base[j * 768 + d]; mn += wt[j] * v[j]; }
        float var = 0.0f;
        for (int j = 0; j < 8; ++j) { float t = v[j] - mn; var += wt[j] * t * t; }
        float mxv = v[0];
        for (int j = 1; j < 8; ++j) mxv = fmaxf(mxv, v[j]);
        P3[orow + d] = mn;
        P3[orow + 768 + d] = sqrtf(var + 1e-8f);
        P3[orow + 1536 + d] = mxv;
    }
}

// ---------------------------------------------------------------------------
// WMMA bf16 GEMM:  C[m,n] = act( sum_k A[m,k]*Wt[k, nw0+n] + bias[bias0+n] ) + res
//   Block tile 64(M) x 256(N), K step 32.
//   8 waves in a 2x4 grid, each wave owns 32x64 = 2x4 WMMA accumulators.
//   B tile is staged global->LDS with GLOBAL_LOAD_ASYNC_TO_LDS_B128 (ASYNCcnt);
//   A tile goes through VGPRs (needs f32->bf16 convert).
// ---------------------------------------------------------------------------
__global__ __launch_bounds__(256) void k_gemm(
    const void* __restrict__ Av, int a_bf16, int M, int K, int lda,
    const __bf16* __restrict__ Wt, int ldw, int nw0,
    const float* __restrict__ bias, int bias0,
    const float* __restrict__ res, int ldr,
    void* __restrict__ Cv, int c_bf16, int ldc, int act)
{
    __shared__ __bf16 lA[64 * 32];    //  4 KB
    __shared__ __bf16 lB[32 * 256];   // 16 KB

    int tid  = threadIdx.x;
    int lane = tid & 31;
    int wid  = tid >> 5;
    int wm   = (wid >> 2) * 32;   // 0 / 32
    int wn   = (wid & 3) * 64;    // 0,64,128,192
    int mtile = blockIdx.y * 64;
    int ntile = blockIdx.x * 256;

    int ar = tid >> 2;            // 0..63
    int ak = (tid & 3) * 8;       // 0,8,16,24
    int bk = tid >> 3;            // 0..31
    int bn = (tid & 7) * 32;      // 0..224

    int mrow = lane & 15;
    int half = lane >> 4;

    const float*  Af = (const float*)Av;
    const __bf16* Ab = (const __bf16*)Av;

    v8f acc[2][4] = {};

    union Frag { v16bf v; uint4 u[2]; };

    for (int k0 = 0; k0 < K; k0 += 32) {
        // ---- stage A tile (64x32) into LDS, converting to bf16 if needed
        union { uint4 u; __bf16 h[8]; } pa;
        int m = mtile + ar;
        if (m < M) {
            if (a_bf16) {
                pa.u = *(const uint4*)(Ab + (size_t)m * lda + k0 + ak);
            } else {
                const float* p = Af + (size_t)m * lda + k0 + ak;
                float4 f0 = *(const float4*)p;
                float4 f1 = *(const float4*)(p + 4);
                pa.h[0] = f2bf(f0.x); pa.h[1] = f2bf(f0.y);
                pa.h[2] = f2bf(f0.z); pa.h[3] = f2bf(f0.w);
                pa.h[4] = f2bf(f1.x); pa.h[5] = f2bf(f1.y);
                pa.h[6] = f2bf(f1.z); pa.h[7] = f2bf(f1.w);
            }
            if (k0 + 32 < K)
                __builtin_prefetch((a_bf16 ? (const void*)(Ab + (size_t)m * lda + k0 + 32 + ak)
                                           : (const void*)(Af + (size_t)m * lda + k0 + 32 + ak)), 0, 1);
        } else {
            pa.u = make_uint4(0u, 0u, 0u, 0u);
        }
        *(uint4*)&lA[ar * 32 + ak] = pa.u;

        // ---- stage B tile (32x256) into LDS via async copy (no VGPR round-trip)
        {
            const __bf16* gsrc = Wt + (size_t)(k0 + bk) * ldw + nw0 + ntile + bn;
#if USE_ASYNC_LDS
            unsigned lb0 = (unsigned)(uintptr_t)&lB[bk * 256 + bn];
#pragma unroll
            for (int c = 0; c < 4; ++c) {
                unsigned long long ga = (unsigned long long)(uintptr_t)(gsrc + c * 8);
                unsigned la = lb0 + c * 16;
                asm volatile("global_load_async_to_lds_b128 %0, %1, off"
                             :: "v"(la), "v"(ga) : "memory");
            }
#else
            uint4* dst = (uint4*)&lB[bk * 256 + bn];
#pragma unroll
            for (int c = 0; c < 4; ++c)
                dst[c] = *(const uint4*)(gsrc + c * 8);
#endif
        }
#if USE_ASYNC_LDS
        asm volatile("s_wait_asynccnt 0" ::: "memory");
#endif
        __syncthreads();

        // ---- build fragments and issue 8 WMMAs (2 A-frags x 4 B-frags)
        Frag afr[2], bfr[4];
#pragma unroll
        for (int i = 0; i < 2; ++i) {
            const __bf16* ap = &lA[(wm + i * 16 + mrow) * 32];
            afr[i].u[0] = *(const uint4*)(ap + half * 8);
            afr[i].u[1] = *(const uint4*)(ap + 16 + half * 8);
        }
#pragma unroll
        for (int j = 0; j < 4; ++j) {
            const __bf16* bq = &lB[wn + j * 16 + mrow];
#pragma unroll
            for (int s = 0; s < 8; ++s) {
                bfr[j].v[s]     = bq[(half * 8 + s) * 256];
                bfr[j].v[8 + s] = bq[(16 + half * 8 + s) * 256];
            }
        }
#pragma unroll
        for (int i = 0; i < 2; ++i)
#pragma unroll
            for (int j = 0; j < 4; ++j)
                acc[i][j] = __builtin_amdgcn_wmma_f32_16x16x32_bf16(
                    false, afr[i].v, false, bfr[j].v, (short)0, acc[i][j], false, false);

        __syncthreads();
    }

    // ---- epilogue: C/D layout is lane(0-15)->N, vgpr r -> M (+8 for upper half)
#pragma unroll
    for (int i = 0; i < 2; ++i)
#pragma unroll
        for (int j = 0; j < 4; ++j) {
            int n = ntile + wn + j * 16 + mrow;
#pragma unroll
            for (int r = 0; r < 8; ++r) {
                int m = mtile + wm + i * 16 + half * 8 + r;
                if (m < M) {
                    float v = acc[i][j][r] + bias[bias0 + n];
                    if (act == 1) v = gelu_f(v);
                    if (res) v += res[(size_t)m * ldr + n];
                    if (c_bf16) ((__bf16*)Cv)[(size_t)m * ldc + n] = f2bf(v);
                    else        ((float*)Cv)[(size_t)m * ldc + n] = v;
                }
            }
        }
}

// ---------------------------------------------------------------------------
// st[b,d] = mean over 512 blocks of pm[b,m,d]
// ---------------------------------------------------------------------------
__global__ __launch_bounds__(256) void k_mean(const float* __restrict__ pm,
                                              float* __restrict__ st) {
    int i = blockIdx.x * 256 + threadIdx.x;
    if (i >= 3072) return;
    int b = i / 768, d = i % 768;
    float s = 0.0f;
    for (int m = 0; m < 512; ++m) s += pm[((size_t)(b * 512 + m)) * 768 + d];
    st[i] = s * (1.0f / 512.0f);
}

// ---------------------------------------------------------------------------
// scores s[b,m] = dot(q[b], k[b,m]) / sqrt(768)  — one wave per (b,m)
// ---------------------------------------------------------------------------
__global__ __launch_bounds__(256) void k_scores(const float* __restrict__ qc,
                                                const float* __restrict__ kc,
                                                float* __restrict__ sc) {
    int gw = blockIdx.x * 8 + (threadIdx.x >> 5);
    int lane = threadIdx.x & 31;
    if (gw >= 2048) return;
    int b = gw >> 9;
    const float* q = qc + b * 768;
    const float* k = kc + (size_t)gw * 768;
    float s = 0.0f;
    for (int d = lane; d < 768; d += 32) s += q[d] * k[d];
    for (int o = 16; o > 0; o >>= 1) s += __shfl_xor(s, o);
    if (lane == 0) sc[gw] = s * 0.03608439182435161f;  // 1/sqrt(768)
}

// ---------------------------------------------------------------------------
// argmax over 512 scores per batch -> clipped window start
// ---------------------------------------------------------------------------
__global__ __launch_bounds__(256) void k_argmax(const float* __restrict__ sc,
                                                int* __restrict__ startb) {
    int b = blockIdx.x, tid = threadIdx.x;
    __shared__ float bv[256];
    __shared__ int   bi[256];
    float best = -3.0e38f; int besti = 0x7FFFFFFF;
    for (int m = tid; m < 512; m += 256) {
        float v = sc[b * 512 + m];
        if (v > best || (v == best && m < besti)) { best = v; besti = m; }
    }
    bv[tid] = best; bi[tid] = besti;
    __syncthreads();
    for (int s = 128; s > 0; s >>= 1) {
        if (tid < s) {
            if (bv[tid + s] > bv[tid] || (bv[tid + s] == bv[tid] && bi[tid + s] < bi[tid])) {
                bv[tid] = bv[tid + s]; bi[tid] = bi[tid + s];
            }
        }
        __syncthreads();
    }
    if (tid == 0) {
        int fp = bi[0] * 8 - 32;
        if (fp < 0) fp = 0;
        if (fp > 4096 - 64) fp = 4096 - 64;
        startb[b] = fp;
    }
}

// ---------------------------------------------------------------------------
// gather window rows of h into ac at slot t64
// ---------------------------------------------------------------------------
__global__ __launch_bounds__(256) void k_gather(const float* __restrict__ h,
                                                const int* __restrict__ startb,
                                                float* __restrict__ ac, int t64) {
    int rid = blockIdx.x;            // 0..255
    int b = rid >> 6, j = rid & 63;
    int s = startb[b];
    const float* src = h + ((size_t)(b * 4096 + s + j)) * 768;
    float* dst = ac + ((size_t)(b * 256 + t64 + j)) * 768;
    for (int d = threadIdx.x; d < 768; d += 256) dst[d] = src[d];
}

// ctx[b,0]=st[b]; ctx[b,1+i]=ac[b,i]  (ctx batch stride = 257 rows)
__global__ __launch_bounds__(256) void k_ctx(const float* __restrict__ st,
                                             const float* __restrict__ ac,
                                             float* __restrict__ ctx, int Lc) {
    int rid = blockIdx.x;
    int b = rid / Lc, i = rid % Lc;
    float* dst = ctx + ((size_t)(b * 257 + i)) * 768;
    const float* src = (i == 0) ? (st + b * 768)
                                : (ac + ((size_t)(b * 256 + (i - 1))) * 768);
    for (int d = threadIdx.x; d < 768; d += 256) dst[d] = src[d];
}

__global__ __launch_bounds__(256) void k_copycl(const float* __restrict__ ctx,
                                                float* __restrict__ cl) {
    int b = blockIdx.x;
    for (int d = threadIdx.x; d < 768; d += 256)
        cl[b * 768 + d] = ctx[(size_t)b * 257 * 768 + d];
}

// ---------------------------------------------------------------------------
// Generic 12-head attention, head_dim 64, wave per query row.
// ---------------------------------------------------------------------------
__global__ __launch_bounds__(256) void k_attn(
    const float* __restrict__ q, int q_bs, int q_rs, int q_off,
    const float* __restrict__ kmat, const float* __restrict__ vmat,
    int kv_bs, int kv_rs, int k_off, int v_off,
    float* __restrict__ out, int o_bs, int o_rs,
    int Lq, int Lkv)
{
    const float scale = 0.125f;  // 1/sqrt(64)
    int bh = blockIdx.x;
    int b = bh / 12, hh = bh % 12;
    int w = threadIdx.x >> 5, lane = threadIdx.x & 31;
    int qi = blockIdx.y * 8 + w;
    __shared__ float probs[8][288];
    __shared__ float qrow[8][64];
    bool act = (qi < Lq);
    if (act) {
        const float* qp = q + ((size_t)(b * q_bs + qi)) * q_rs + q_off + hh * 64;
        qrow[w][lane] = qp[lane];
        qrow[w][lane + 32] = qp[lane + 32];
    }
    __syncthreads();
    if (act) {
        for (int k = lane; k < Lkv; k += 32) {
            const float* kp = kmat + ((size_t)(b * kv_bs + k)) * kv_rs + k_off + hh * 64;
            float s = 0.0f;
            for (int d = 0; d < 64; ++d) s += qrow[w][d] * kp[d];
            probs[w][k] = s * scale;
        }
    }
    __syncthreads();
    float mx = -3.0e38f, sum = 0.0f;
    if (act) for (int k = lane; k < Lkv; k += 32) mx = fmaxf(mx, probs[w][k]);
    for (int o = 16; o > 0; o >>= 1) mx = fmaxf(mx, __shfl_xor(mx, o));
    if (act) for (int k = lane; k < Lkv; k += 32) {
        float e = expf(probs[w][k] - mx);
        probs[w][k] = e;
        sum += e;
    }
    for (int o = 16; o > 0; o >>= 1) sum += __shfl_xor(sum, o);
    __syncthreads();
    if (act) {
        float inv = 1.0f / sum;
        float a0 = 0.0f, a1 = 0.0f;
        for (int k = 0; k < Lkv; ++k) {
            const float* vp = vmat + ((size_t)(b * kv_bs + k)) * kv_rs + v_off + hh * 64;
            float p = probs[w][k];
            a0 += p * vp[lane];
            a1 += p * vp[lane + 32];
        }
        float* op = out + ((size_t)(b * o_bs + qi)) * o_rs + hh * 64;
        op[lane] = a0 * inv;
        op[lane + 32] = a1 * inv;
    }
}

// ---------------------------------------------------------------------------
// gate scalar a = sigmoid( gelu(t*w1 + b1) . w2 + b2 )
// ---------------------------------------------------------------------------
__global__ __launch_bounds__(256) void k_gate(const float* __restrict__ w1,
                                              const float* __restrict__ b1,
                                              const float* __restrict__ w2,
                                              const float* __restrict__ b2,
                                              float tval, float* __restrict__ a_out) {
    __shared__ float red[256];
    int tid = threadIdx.x;
    float s = 0.0f;
    for (int d = tid; d < 768; d += 256)
        s += gelu_f(tval * w1[d] + b1[d]) * w2[d];
    red[tid] = s;
    __syncthreads();
    for (int st = 128; st > 0; st >>= 1) { if (tid < st) red[tid] += red[tid + st]; __syncthreads(); }
    if (tid == 0) a_out[0] = 1.0f / (1.0f + expf(-(red[0] + b2[0])));
}

__global__ __launch_bounds__(256) void k_axpy(float* __restrict__ pm,
                                              const float* __restrict__ d,
                                              const float* __restrict__ a, int n) {
    int i = blockIdx.x * 256 + threadIdx.x;
    if (i < n) pm[i] += a[0] * d[i];
}

// out[b,n,:] = x[b,n,:] + y[b,:]
__global__ __launch_bounds__(256) void k_bcast(const float* __restrict__ x,
                                               const float* __restrict__ y,
                                               float* __restrict__ out) {
    size_t i = (size_t)blockIdx.x * 256 + threadIdx.x;
    int d = (int)(i % 768);
    int b = (int)(i / (4096u * 768u));
    out[i] = x[i] + y[b * 768 + d];
}

// ---------------------------------------------------------------------------
// host orchestration
// ---------------------------------------------------------------------------
extern "C" void kernel_launch(void* const* d_in, const int* in_sizes, int n_in,
                              void* d_out, int out_size, void* d_ws, size_t ws_size,
                              hipStream_t stream) {
    (void)in_sizes; (void)n_in; (void)out_size; (void)ws_size;

    const float* x        = (const float*)d_in[0];
    const float* ln1_g    = (const float*)d_in[1];
    const float* ln1_b    = (const float*)d_in[2];
    const float* ln2_g    = (const float*)d_in[3];
    const float* ln2_b    = (const float*)d_in[4];
    const float* mlp_w1   = (const float*)d_in[5];
    const float* mlp_b1   = (const float*)d_in[6];
    const float* mlp_w2   = (const float*)d_in[7];
    const float* mlp_b2   = (const float*)d_in[8];
    const float* pe_ww    = (const float*)d_in[9];
    const float* pe_wb    = (const float*)d_in[10];
    const float* pe_proj_w= (const float*)d_in[11];
    const float* pe_proj_b= (const float*)d_in[12];
    const float* pe_ng    = (const float*)d_in[13];
    const float* pe_nb    = (const float*)d_in[14];
    const float* pe_pos   = (const float*)d_in[15];
    const float* ctrl_qw  = (const float*)d_in[16];
    const float* ctrl_qb  = (const float*)d_in[17];
    const float* ctrl_kw  = (const float*)d_in[18];
    const float* ctrl_kb  = (const float*)d_in[19];
    const float* fov_in_w = (const float*)d_in[20];
    const float* fov_in_b = (const float*)d_in[21];
    const float* fov_out_w= (const float*)d_in[22];
    const float* fov_out_b= (const float*)d_in[23];
    const float* fov_n1g  = (const float*)d_in[24];
    const float* fov_n1b  = (const float*)d_in[25];
    const float* fov_n2g  = (const float*)d_in[26];
    const float* fov_n2b  = (const float*)d_in[27];
    const float* fov_ff_w1= (const float*)d_in[28];
    const float* fov_ff_b1= (const float*)d_in[29];
    const float* fov_ff_w2= (const float*)d_in[30];
    const float* fov_ff_b2= (const float*)d_in[31];
    const float* ma_in_w  = (const float*)d_in[32];
    const float* ma_in_b  = (const float*)d_in[33];
    const float* ma_out_w = (const float*)d_in[34];
    const float* ma_out_b = (const float*)d_in[35];
    const float* mn_g     = (const float*)d_in[36];
    const float* mn_b     = (const float*)d_in[37];
    const float* mg_w1    = (const float*)d_in[38];
    const float* mg_b1    = (const float*)d_in[39];
    const float* mg_w2    = (const float*)d_in[40];
    const float* mg_b2    = (const float*)d_in[41];
    const float* op_w     = (const float*)d_in[42];
    const float* op_b     = (const float*)d_in[43];
    const float* on_g     = (const float*)d_in[44];
    const float* on_b     = (const float*)d_in[45];

    // deterministic bump allocator over d_ws
    size_t off = 0;
    auto alloc = [&](size_t nbytes) -> void* {
        void* p = (char*)d_ws + off;
        off += (nbytes + 255) & ~(size_t)255;
        return p;
    };

    __bf16* Wt_pe = (__bf16*)alloc(2304ull * 768 * 2);
    __bf16* Wt_cq = (__bf16*)alloc(768ull * 768 * 2);
    __bf16* Wt_ck = (__bf16*)alloc(768ull * 768 * 2);
    __bf16* Wt_fi = (__bf16*)alloc(768ull * 2304 * 2);
    __bf16* Wt_fo = (__bf16*)alloc(768ull * 768 * 2);
    __bf16* Wt_f1 = (__bf16*)alloc(768ull * 3072 * 2);
    __bf16* Wt_f2 = (__bf16*)alloc(3072ull * 768 * 2);
    __bf16* Wt_mi = (__bf16*)alloc(768ull * 2304 * 2);
    __bf16* Wt_mo = (__bf16*)alloc(768ull * 768 * 2);
    __bf16* Wt_opw= (__bf16*)alloc(768ull * 768 * 2);
    __bf16* Wt_m1 = (__bf16*)alloc(768ull * 3072 * 2);
    __bf16* Wt_m2 = (__bf16*)alloc(3072ull * 768 * 2);

    float* h    = (float*)alloc(16384ull * 768 * 4);
    float* P3   = (float*)alloc(2048ull * 2304 * 4);
    float* pm   = (float*)alloc(2048ull * 768 * 4);
    float* pmn  = (float*)alloc(2048ull * 768 * 4);
    float* stb  = (float*)alloc(4ull * 768 * 4);
    float* qc   = (float*)alloc(4ull * 768 * 4);
    float* kc   = (float*)alloc(2048ull * 768 * 4);
    float* sc   = (float*)alloc(2048ull * 4);
    int*   startb = (int*)alloc(16);
    float* ac   = (float*)alloc(1024ull * 768 * 4);
    float* ctx  = (float*)alloc(1028ull * 768 * 4);
    float* nctx = (float*)alloc(1028ull * 768 * 4);
    float* qkvf = (float*)alloc(1028ull * 2304 * 4);
    float* aof  = (float*)alloc(1028ull * 768 * 4);
    float* cl   = (float*)alloc(4ull * 768 * 4);
    float* cl2  = (float*)alloc(4ull * 768 * 4);
    float* f1b  = (float*)alloc(4ull * 3072 * 4);
    float* qma  = (float*)alloc(2048ull * 768 * 4);
    float* kvm  = (float*)alloc(1024ull * 1536 * 4);
    float* aom  = (float*)alloc(2048ull * 768 * 4);
    float* ddm  = (float*)alloc(2048ull * 768 * 4);
    float* agate= (float*)alloc(256);
    float* stn  = (float*)alloc(4ull * 768 * 4);
    float* ybuf = (float*)alloc(4ull * 768 * 4);
    __bf16* t1  = (__bf16*)alloc(16384ull * 3072 * 2);
    float* x2n  = h;  // reuse: h is dead after the saccade loop

    auto w2bf = [&](const float* W, __bf16* Wt, int Nout, int K) {
        size_t n = (size_t)Nout * K;
        k_w2bf<<<dim3((unsigned)((n + 255) / 256)), 256, 0, stream>>>(W, Wt, Nout, K);
    };
    auto gemm = [&](const void* A, int abf, int M, int K, int lda,
                    const __bf16* Wt, int ldw, int nw0,
                    const float* bias, int bias0,
                    const float* res, int ldr,
                    void* C, int cbf, int ldc, int act, int Ncols) {
        dim3 grid(Ncols / 256, (M + 63) / 64);
        k_gemm<<<grid, 256, 0, stream>>>(A, abf, M, K, lda, Wt, ldw, nw0,
                                         bias, bias0, res, ldr, C, cbf, ldc, act);
    };

    // ---- weight conversion (bf16, transposed)
    w2bf(pe_proj_w, Wt_pe, 768, 2304);
    w2bf(ctrl_qw,   Wt_cq, 768, 768);
    w2bf(ctrl_kw,   Wt_ck, 768, 768);
    w2bf(fov_in_w,  Wt_fi, 2304, 768);
    w2bf(fov_out_w, Wt_fo, 768, 768);
    w2bf(fov_ff_w1, Wt_f1, 3072, 768);
    w2bf(fov_ff_w2, Wt_f2, 768, 3072);
    w2bf(ma_in_w,   Wt_mi, 2304, 768);
    w2bf(ma_out_w,  Wt_mo, 768, 768);
    w2bf(op_w,      Wt_opw, 768, 768);
    w2bf(mlp_w1,    Wt_m1, 3072, 768);
    w2bf(mlp_w2,    Wt_m2, 768, 3072);

    // ---- h = LN(x)
    k_ln<<<16384, 256, 0, stream>>>(x, h, ln1_g, ln1_b, nullptr, 1);

    // ---- pooled block stats -> P3 ; pm = LN(P3 @ pe_proj^T + b) + pos
    k_pool<<<2048, 256, 0, stream>>>(h, pe_ww, pe_wb, P3);
    gemm(P3, 0, 2048, 2304, 2304, Wt_pe, 768, 0, pe_proj_b, 0, nullptr, 0, pm, 0, 768, 0, 768);
    k_ln<<<2048, 256, 0, stream>>>(pm, pm, pe_ng, pe_nb, pe_pos, 512);
    k_mean<<<12, 256, 0, stream>>>(pm, stb);

    // ---- NS saccade steps
    for (int t = 0; t < 4; ++t) {
        int La = 64 * (t + 1);
        int Lc = 1 + La;

        gemm(stb, 0, 4, 768, 768, Wt_cq, 768, 0, ctrl_qb, 0, nullptr, 0, qc, 0, 768, 0, 768);
        gemm(pm, 0, 2048, 768, 768, Wt_ck, 768, 0, ctrl_kb, 0, nullptr, 0, kc, 0, 768, 0, 768);
        k_scores<<<256, 256, 0, stream>>>(qc, kc, sc);
        k_argmax<<<4, 256, 0, stream>>>(sc, startb);
        k_gather<<<256, 256, 0, stream>>>(h, startb, ac, t * 64);
        k_ctx<<<4 * Lc, 256, 0, stream>>>(stb, ac, ctx, Lc);

        // fovea MHA on ctx (self-attention)
        k_ln<<<1028, 256, 0, stream>>>(ctx, nctx, fov_n1g, fov_n1b, nullptr, 1);
        gemm(nctx, 0, 1028, 768, 768, Wt_fi, 2304, 0, fov_in_b, 0, nullptr, 0, qkvf, 0, 2304, 0, 2304);
        k_attn<<<dim3(48, (Lc + 7) / 8), 256, 0, stream>>>(
            qkvf, 257, 2304, 0, qkvf, qkvf, 257, 2304, 768, 1536, aof, 257, 768, Lc, Lc);
        gemm(aof, 0, 1028, 768, 768, Wt_fo, 768, 0, fov_out_b, 0, ctx, 768, ctx, 0, 768, 0, 768);

        // CLS feed-forward -> new st
        k_copycl<<<4, 256, 0, stream>>>(ctx, cl);
        k_ln<<<4, 256, 0, stream>>>(cl, cl2, fov_n2g, fov_n2b, nullptr, 1);
        gemm(cl2, 0, 4, 768, 768, Wt_f1, 3072, 0, fov_ff_b1, 0, nullptr, 0, f1b, 0, 3072, 1, 3072);
        gemm(f1b, 0, 4, 3072, 3072, Wt_f2, 768, 0, fov_ff_b2, 0, cl, 768, stb, 0, 768, 0, 768);

        // memory cross-attention pm <- pm + a * MHA(LN(pm), ac, ac)
        k_ln<<<2048, 256, 0, stream>>>(pm, pmn, mn_g, mn_b, nullptr, 1);
        gemm(pmn, 0, 2048, 768, 768, Wt_mi, 2304, 0, ma_in_b, 0, nullptr, 0, qma, 0, 768, 0, 768);
        gemm(ac, 0, 1024, 768, 768, Wt_mi, 2304, 768, ma_in_b, 768, nullptr, 0, kvm, 0, 1536, 0, 1536);
        k_attn<<<dim3(48, 64), 256, 0, stream>>>(
            qma, 512, 768, 0, kvm, kvm, 256, 1536, 0, 768, aom, 512, 768, 512, La);
        gemm(aom, 0, 2048, 768, 768, Wt_mo, 768, 0, ma_out_b, 0, nullptr, 0, ddm, 0, 768, 0, 768);
        k_gate<<<1, 256, 0, stream>>>(mg_w1, mg_b1, mg_w2, mg_b2, (float)t / 4.0f, agate);
        k_axpy<<<(2048 * 768 + 255) / 256, 256, 0, stream>>>(pm, ddm, agate, 2048 * 768);
    }

    // ---- output head + broadcast residual
    k_ln<<<4, 256, 0, stream>>>(stb, stn, on_g, on_b, nullptr, 1);
    gemm(stn, 0, 4, 768, 768, Wt_opw, 768, 0, op_b, 0, nullptr, 0, ybuf, 0, 768, 0, 768);
    k_bcast<<<49152, 256, 0, stream>>>(x, ybuf, (float*)d_out);

    // ---- big MLP: d_out += gelu(LN(d_out) @ W1^T + b1) @ W2^T + b2
    k_ln<<<16384, 256, 0, stream>>>((const float*)d_out, x2n, ln2_g, ln2_b, nullptr, 1);
    gemm(x2n, 0, 16384, 768, 768, Wt_m1, 3072, 0, mlp_b1, 0, nullptr, 0, t1, 1, 3072, 1, 3072);
    gemm(t1, 1, 16384, 3072, 3072, Wt_m2, 768, 0, mlp_b2, 0, (const float*)d_out, 768, d_out, 0, 768, 0, 768);
}
